// NodeSchNetWrapper_12180527252068
// MI455X (gfx1250) — compile-verified, matrix operand order checked by reading, not run on hardware
//
#include <hip/hip_runtime.h>
#include <hip/hip_bf16.h>
#include <math.h>

// ---------------------------------------------------------------------------
// SchNet forward, CDNA5 (gfx1250) wave32 + bf16 WMMA implementation.
// ---------------------------------------------------------------------------

typedef __bf16 bf16;
typedef bf16 v8bf  __attribute__((ext_vector_type(8)));
typedef bf16 v16bf __attribute__((ext_vector_type(16)));
typedef float v8f  __attribute__((ext_vector_type(8)));

#define B_GRAPHS 64
#define NA       32
#define N_TOT    (B_GRAPHS * NA)   // 2048 atoms
#define HID      600
#define HP       608               // padded hidden (38 * 16, 19 * 32)
#define NG       50
#define NGP      64                // padded gaussians
#define NLAYERS  6
#define NEDGE    (N_TOT * NA)      // 65536 block-dense edges
#define NI       2                 // atoms per cfconv workgroup (4 M-tiles / B fetch)
#define CUT      10.0f
#define LN2      0.6931471805599453f

__device__ __forceinline__ float sspf(float x) {
    // shifted softplus: log(1+e^x) - log 2   (ssp(0) == 0, keeps padding exact)
    float sp = (x > 20.f) ? x : log1pf(expf(x));
    return sp - LN2;
}

// Load a 16-element bf16 WMMA fragment for one lane.
// CDNA5 16-bit A/B layout: lanes 0-15 hold K {k0..k0+7, k0+16..k0+23},
// lanes 16-31 hold K {k0+8..k0+15, k0+24..k0+31}  (ISA 7.12.2).
__device__ __forceinline__ v16bf load_frag(const bf16* __restrict__ base, int half) {
    v8bf lo = *(const v8bf*)(base + half);
    v8bf hi = *(const v8bf*)(base + half + 16);
    v16bf f;
#pragma unroll
    for (int k = 0; k < 8; ++k) { f[k] = lo[k]; f[8 + k] = hi[k]; }
    return f;
}

__device__ __forceinline__ v8f wmma_bf16(v16bf a, v16bf b, v8f c) {
    return __builtin_amdgcn_wmma_f32_16x16x32_bf16(false, a, false, b, (short)0, c,
                                                   false, false);
}

// ---------------------------------------------------------------------------
// Weight prep: fp32 [K,N] row-major  ->  bf16 [Np, Kp] (transposed, zero pad)
// ---------------------------------------------------------------------------
__global__ void transpose_pad_kernel(const float* __restrict__ src,
                                     bf16* __restrict__ dst,
                                     int K, int N, int Kp, int Np) {
    int idx = blockIdx.x * blockDim.x + threadIdx.x;
    if (idx >= Np * Kp) return;
    int n = idx / Kp, k = idx % Kp;
    float v = (k < K && n < N) ? src[k * N + n] : 0.f;
    dst[idx] = (bf16)v;
}

// ---------------------------------------------------------------------------
// Geometry: gaussian smearing (bf16, padded to 64) + masked cosine cutoff
// ---------------------------------------------------------------------------
__global__ void geom_kernel(const float* __restrict__ pos,
                            bf16* __restrict__ rbf,
                            float* __restrict__ cosw) {
    int idx = blockIdx.x * blockDim.x + threadIdx.x;  // edge*64 + g
    if (idx >= NEDGE * NGP) return;
    int e = idx >> 6, g = idx & 63;
    int b = e >> 10, i = (e >> 5) & 31, j = e & 31;
    int ai = b * NA + i, aj = b * NA + j;
    float dx = pos[ai * 3 + 0] - pos[aj * 3 + 0];
    float dy = pos[ai * 3 + 1] - pos[aj * 3 + 1];
    float dz = pos[ai * 3 + 2] - pos[aj * 3 + 2];
    float sq = dx * dx + dy * dy + dz * dz;
    float d  = sqrtf(sq > 0.f ? sq : 1.f);
    const float spacing = CUT / (float)(NG - 1);
    const float coeff   = -0.5f / (spacing * spacing);
    float r = 0.f;
    if (g < NG) { float t = d - spacing * (float)g; r = expf(coeff * t * t); }
    rbf[(long)e * NGP + g] = (bf16)r;
    if (g == 0) {
        bool m = (sq <= CUT * CUT) && (i != j);
        cosw[e] = m ? 0.5f * (cosf(d * 3.14159265358979323846f / CUT) + 1.f) : 0.f;
    }
}

// ---------------------------------------------------------------------------
// Embedding gather h = emb[z], zero padded to HP columns
// ---------------------------------------------------------------------------
__global__ void embed_kernel(const int* __restrict__ z,
                             const float* __restrict__ emb,
                             float* __restrict__ h) {
    int idx = blockIdx.x * blockDim.x + threadIdx.x;
    if (idx >= N_TOT * HP) return;
    int a = idx / HP, n = idx % HP;
    h[idx] = (n < HID) ? emb[z[a] * HID + n] : 0.f;
}

// ---------------------------------------------------------------------------
// Generic WMMA linear: Y = [res +] act(X @ WT^T + bias)
//   X  fp32 [M, xpitch], WT bf16 [HP, HP] (N-major, K padded)
//   one workgroup = 16 rows of M, 8 waves split the 38 N-tiles
// ---------------------------------------------------------------------------
__global__ void __launch_bounds__(256)
lin_kernel(const float* __restrict__ X, int xpitch,
           const bf16* __restrict__ WT,
           const float* __restrict__ bias,
           const float* __restrict__ res, int respitch,
           float* __restrict__ Y, int ypitch, int nlimit, int act) {
    __shared__ bf16 xa[16 * HP];
    int tid = threadIdx.x;
    int lane = tid & 31, wave = tid >> 5;
    int m0 = blockIdx.x * 16;

    for (int p = tid; p < 16 * HP; p += 256) {
        int r = p / HP, c = p % HP;
        xa[p] = (bf16)X[(long)(m0 + r) * xpitch + c];
    }
    __syncthreads();

    const int half = (lane < 16) ? 0 : 8;
    const bf16* arow = xa + (lane & 15) * HP;

    for (int nt = wave; nt < HP / 16; nt += 8) {
        int n = nt * 16 + (lane & 15);
        const bf16* wrow = WT + (long)n * HP;
        // warm L2/WGP$ for this wave's next N-tile panel (global_prefetch_b8)
        if (nt + 8 < HP / 16) __builtin_prefetch(wrow + (size_t)8 * 16 * HP, 0, 0);
        v8f acc = {};
        for (int ks = 0; ks < HP / 32; ++ks) {
            v16bf af = load_frag(arow + ks * 32, half);
            v16bf bf_ = load_frag(wrow + ks * 32, half);
            acc = wmma_bf16(af, bf_, acc);
        }
        float bn = (bias != nullptr && n < HID) ? bias[n] : 0.f;
#pragma unroll
        for (int r = 0; r < 8; ++r) {
            int m = m0 + r + ((lane < 16) ? 0 : 8);
            float y = acc[r] + bn;
            if (act) y = sspf(y);
            if (res != nullptr) y += res[(long)m * respitch + n];
            if (n < nlimit) Y[(long)m * ypitch + n] = y;
        }
    }
}

// ---------------------------------------------------------------------------
// Fused CFConv: one workgroup per NI atoms of one graph (NI*32 edge rows).
//   hidden = ssp(rbf[NI*32,64] @ w1T + b1)           (bf16 in LDS, never HBM)
//   W      = hidden @ w2T + b2                       (registers only)
//   m[i,:] = sum_j  W[j,:] * cosw[j] * xj[j,:]       (in-register reduce)
// Each B-fragment fetch feeds 2*NI WMMAs -> halved L2 weight traffic vs NI=1.
// xj tile is per-graph and staged once for both atoms.
// ---------------------------------------------------------------------------
__global__ void __launch_bounds__(256)
cfconv_kernel(const bf16* __restrict__ rbf, const float* __restrict__ cosw,
              const float* __restrict__ xj,
              const bf16* __restrict__ w1T, const float* __restrict__ b1,
              const bf16* __restrict__ w2T, const float* __restrict__ b2,
              float* __restrict__ mout) {
    extern __shared__ char smem[];
    bf16*  rbf_s = (bf16*)smem;                                   // NI*32 x 64
    bf16*  hid_s = (bf16*)(smem + NI * 32 * NGP * 2);             // NI*32 x HP
    bf16*  xj_s  = (bf16*)(smem + NI * 32 * NGP * 2 + NI * 32 * HP * 2); // 32 x HP
    float* cw_s  = (float*)(smem + NI * 32 * NGP * 2 + NI * 32 * HP * 2 + 32 * HP * 2);

    int tid = threadIdx.x, lane = tid & 31, wave = tid >> 5;
    int blk = blockIdx.x;                 // 0 .. N_TOT/NI - 1
    int b  = blk / (NA / NI);
    int i0 = (blk % (NA / NI)) * NI;      // first atom handled here
    int abase = b * NA;
    long e0 = ((long)abase + i0) * NA;    // NI*32 consecutive edge rows

    { // stage rbf tiles (vectorized) + cutoff weights
        const unsigned* rs = (const unsigned*)(rbf + e0 * NGP);
        unsigned* rd = (unsigned*)rbf_s;
        for (int p = tid; p < NI * 32 * NGP / 2; p += 256) rd[p] = rs[p];
        if (tid < NI * NA) cw_s[tid] = cosw[e0 + tid];
    }
    // stage xj rows of this graph's 32 atoms (shared by both i), fp32 -> bf16
    for (int p = tid; p < 32 * HP; p += 256) {
        int j = p / HP, c = p % HP;
        xj_s[p] = (bf16)xj[(long)(abase + j) * HP + c];
    }
    __syncthreads();

    const int half = (lane < 16) ? 0 : 8;

    // ---- GEMM1: hidden[NI*32, HP] = ssp(rbf @ w1T + b1), K = 64 -----------
    for (int t = wave; t < (HP / 16) * (NI * 2); t += 8) {
        int nt = t / (NI * 2), mt = t % (NI * 2);
        int n = nt * 16 + (lane & 15);
        const bf16* arow = rbf_s + (mt * 16 + (lane & 15)) * NGP;
        const bf16* wrow = w1T + (long)n * NGP;
        v8f acc = {};
#pragma unroll
        for (int ks = 0; ks < 2; ++ks) {
            v16bf af = load_frag(arow + ks * 32, half);
            v16bf bf_ = load_frag(wrow + ks * 32, half);
            acc = wmma_bf16(af, bf_, acc);
        }
        float bn = (n < HID) ? b1[n] : 0.f;
#pragma unroll
        for (int r = 0; r < 8; ++r) {
            int row = mt * 16 + r + ((lane < 16) ? 0 : 8);
            hid_s[row * HP + n] = (bf16)sspf(acc[r] + bn);
        }
    }
    __syncthreads();

    // ---- GEMM2 + neighbor reduction, K = 608 ------------------------------
    for (int nt = wave; nt < HP / 16; nt += 8) {
        int n = nt * 16 + (lane & 15);
        const bf16* wrow = w2T + (long)n * HP;
        if (nt + 8 < HP / 16) __builtin_prefetch(wrow + (size_t)8 * 16 * HP, 0, 0);
        v8f acc[NI * 2];
#pragma unroll
        for (int mt = 0; mt < NI * 2; ++mt) acc[mt] = (v8f){};
#pragma unroll 2
        for (int ks = 0; ks < HP / 32; ++ks) {
            v16bf bf_ = load_frag(wrow + ks * 32, half);
#pragma unroll
            for (int mt = 0; mt < NI * 2; ++mt) {
                v16bf a = load_frag(hid_s + (mt * 16 + (lane & 15)) * HP + ks * 32, half);
                acc[mt] = wmma_bf16(a, bf_, acc[mt]);
            }
        }
        float bn = (n < HID) ? b2[n] : 0.f;
#pragma unroll
        for (int a = 0; a < NI; ++a) {
            float sum = 0.f;
#pragma unroll
            for (int r = 0; r < 8; ++r) {
                int j0 = r + ((lane < 16) ? 0 : 8);
                int j1 = 16 + j0;
                sum += (acc[a * 2 + 0][r] + bn) * cw_s[a * 32 + j0] * (float)xj_s[j0 * HP + n];
                sum += (acc[a * 2 + 1][r] + bn) * cw_s[a * 32 + j1] * (float)xj_s[j1 * HP + n];
            }
            sum += __shfl_xor(sum, 16, 32);   // fold lane halves (wave32)
            if (lane < 16) mout[(long)(abase + i0 + a) * HP + n] = sum;
        }
    }
}

// ---------------------------------------------------------------------------
// Per-graph mean pooling
// ---------------------------------------------------------------------------
__global__ void pool_kernel(const float* __restrict__ h, float* __restrict__ pooled) {
    int idx = blockIdx.x * blockDim.x + threadIdx.x;
    if (idx >= B_GRAPHS * HP) return;
    int b = idx / HP, n = idx % HP;
    float s = 0.f;
#pragma unroll 4
    for (int i = 0; i < NA; ++i) s += h[(long)(b * NA + i) * HP + n];
    pooled[idx] = s * (1.f / (float)NA);
}

// ---------------------------------------------------------------------------
// Host launcher
// ---------------------------------------------------------------------------
extern "C" void kernel_launch(void* const* d_in, const int* in_sizes, int n_in,
                              void* d_out, int out_size, void* d_ws, size_t ws_size,
                              hipStream_t stream) {
    (void)in_sizes; (void)n_in; (void)out_size; (void)ws_size;
    const int*   z       = (const int*)  d_in[0];
    const float* pos     = (const float*)d_in[1];
    const float* emb     = (const float*)d_in[3];     // d_in[2] = batch (unused)
    const float* mlp_w1  = (const float*)d_in[4];
    const float* mlp_b1  = (const float*)d_in[5];
    const float* mlp_w2  = (const float*)d_in[6];
    const float* mlp_b2  = (const float*)d_in[7];
    const float* lin1_w  = (const float*)d_in[8];
    const float* lin2_w  = (const float*)d_in[9];
    const float* lin2_b  = (const float*)d_in[10];
    const float* lin_w   = (const float*)d_in[11];
    const float* lin_b   = (const float*)d_in[12];
    const float* pool_w  = (const float*)d_in[13];
    const float* pool_b  = (const float*)d_in[14];
    float* out = (float*)d_out;

    // workspace carve-up (all offsets 256B aligned)
    size_t off = 0;
    auto carve = [&](size_t bytes) -> char* {
        char* p = (char*)d_ws + off;
        off += (bytes + 255) & ~(size_t)255;
        return p;
    };
    float* h      = (float*)carve((size_t)N_TOT * HP * 4);
    float* xjb    = (float*)carve((size_t)N_TOT * HP * 4);
    float* mbuf   = (float*)carve((size_t)N_TOT * HP * 4);
    float* tbuf   = (float*)carve((size_t)N_TOT * HP * 4);
    float* pooled = (float*)carve((size_t)B_GRAPHS * HP * 4);
    bf16*  rbf    = (bf16*) carve((size_t)NEDGE * NGP * 2);
    float* cosw   = (float*)carve((size_t)NEDGE * 4);
    bf16*  w1T    = (bf16*) carve((size_t)NLAYERS * HP * NGP * 2);
    bf16*  w2T    = (bf16*) carve((size_t)NLAYERS * HP * HP * 2);
    bf16*  l1T    = (bf16*) carve((size_t)NLAYERS * HP * HP * 2);
    bf16*  l2T    = (bf16*) carve((size_t)NLAYERS * HP * HP * 2);
    bf16*  lwT    = (bf16*) carve((size_t)NLAYERS * HP * HP * 2);
    bf16*  poolT  = (bf16*) carve((size_t)HP * HP * 2);

    const int TB = 256;
    const int gSq = (HP * HP + TB - 1) / TB;   // 600x600 -> 608x608 transpose
    const int gW1 = (HP * NGP + TB - 1) / TB;

    // 1) weight prep (bf16, transposed, padded)
    for (int l = 0; l < NLAYERS; ++l) {
        transpose_pad_kernel<<<gW1, TB, 0, stream>>>(mlp_w1 + (size_t)l * NG * HID,
                                                     w1T + (size_t)l * HP * NGP,
                                                     NG, HID, NGP, HP);
        transpose_pad_kernel<<<gSq, TB, 0, stream>>>(mlp_w2 + (size_t)l * HID * HID,
                                                     w2T + (size_t)l * HP * HP,
                                                     HID, HID, HP, HP);
        transpose_pad_kernel<<<gSq, TB, 0, stream>>>(lin1_w + (size_t)l * HID * HID,
                                                     l1T + (size_t)l * HP * HP,
                                                     HID, HID, HP, HP);
        transpose_pad_kernel<<<gSq, TB, 0, stream>>>(lin2_w + (size_t)l * HID * HID,
                                                     l2T + (size_t)l * HP * HP,
                                                     HID, HID, HP, HP);
        transpose_pad_kernel<<<gSq, TB, 0, stream>>>(lin_w + (size_t)l * HID * HID,
                                                     lwT + (size_t)l * HP * HP,
                                                     HID, HID, HP, HP);
    }
    transpose_pad_kernel<<<gSq, TB, 0, stream>>>(pool_w, poolT, HID, HID, HP, HP);

    // 2) geometry + embedding
    geom_kernel<<<(NEDGE * NGP + TB - 1) / TB, TB, 0, stream>>>(pos, rbf, cosw);
    embed_kernel<<<(N_TOT * HP + TB - 1) / TB, TB, 0, stream>>>(z, emb, h);

    // 3) interaction layers
    const size_t cf_lds = (size_t)NI * 32 * NGP * 2   // rbf tiles
                        + (size_t)NI * 32 * HP * 2    // hidden tiles
                        + (size_t)32 * HP * 2         // xj tile (per graph)
                        + (size_t)NI * 32 * 4;        // cutoff weights
    const int mtiles = N_TOT / 16;  // 128
    for (int l = 0; l < NLAYERS; ++l) {
        // xj = h @ lin1 (no bias)
        lin_kernel<<<mtiles, TB, 0, stream>>>(h, HP, l1T + (size_t)l * HP * HP,
                                              nullptr, nullptr, 0,
                                              xjb, HP, HP, 0);
        // fused filter-net + message aggregation
        cfconv_kernel<<<N_TOT / NI, TB, cf_lds, stream>>>(
            rbf, cosw, xjb,
            w1T + (size_t)l * HP * NGP, mlp_b1 + (size_t)l * HID,
            w2T + (size_t)l * HP * HP,  mlp_b2 + (size_t)l * HID,
            mbuf);
        // t = ssp(m @ lin2 + b)
        lin_kernel<<<mtiles, TB, 0, stream>>>(mbuf, HP, l2T + (size_t)l * HP * HP,
                                              lin2_b + (size_t)l * HID, nullptr, 0,
                                              tbuf, HP, HP, 1);
        // h = h + (t @ lin + b)
        lin_kernel<<<mtiles, TB, 0, stream>>>(tbuf, HP, lwT + (size_t)l * HP * HP,
                                              lin_b + (size_t)l * HID, h, HP,
                                              h, HP, HP, 0);
    }

    // 4) pool + output projection (writes d_out [64, 600])
    pool_kernel<<<(B_GRAPHS * HP + TB - 1) / TB, TB, 0, stream>>>(h, pooled);
    lin_kernel<<<B_GRAPHS / 16, TB, 0, stream>>>(pooled, HP, poolT,
                                                 pool_b, nullptr, 0,
                                                 out, HID, HID, 0);
}
